// GraphSAGE_45664092291593
// MI455X (gfx1250) — compile-verified, matrix operand order checked by reading, not run on hardware
//
#include <hip/hip_runtime.h>
#include <hip/hip_fp16.h>
#include <math.h>

typedef __attribute__((ext_vector_type(16))) _Float16 v16h;
typedef __attribute__((ext_vector_type(8)))  float    v8f;

#define F_IN 128
#define HID  64
#define NCLS 40

// ---------------------------------------------------------------------------
// Utility: zero a float buffer (grid-stride)
// ---------------------------------------------------------------------------
__global__ void zero_f32_kernel(float* __restrict__ p, long long n) {
  long long i = (long long)blockIdx.x * blockDim.x + threadIdx.x;
  long long stride = (long long)gridDim.x * blockDim.x;
  for (; i < n; i += stride) p[i] = 0.0f;
}

// ---------------------------------------------------------------------------
// In-degree counts (float, for mean divide)
// ---------------------------------------------------------------------------
__global__ void degree_kernel(const int* __restrict__ dst,
                              float* __restrict__ cnt, int E) {
  int i = blockIdx.x * blockDim.x + threadIdx.x;
  int stride = gridDim.x * blockDim.x;
  for (; i < E; i += stride) atomicAdd(&cnt[dst[i]], 1.0f);
}

// ---------------------------------------------------------------------------
// Edge scatter-sum: summed[dst] += x[src], float4 gather + f32 atomics.
// x and summed are L2-resident (51MB / 26MB vs 192MB L2) -> bound by L2
// atomic-RMW throughput, the best available path for an unsorted edge list.
// ---------------------------------------------------------------------------
template <int F>
__global__ void scatter_sum_kernel(const float* __restrict__ x,
                                   const int* __restrict__ src,
                                   const int* __restrict__ dst,
                                   float* __restrict__ summed, int E) {
  constexpr int FQ = F / 4;
  long long total = (long long)E * FQ;
  long long i = (long long)blockIdx.x * blockDim.x + threadIdx.x;
  long long stride = (long long)gridDim.x * blockDim.x;
  for (; i < total; i += stride) {
    int e = (int)(i / FQ);
    int g = (int)(i - (long long)e * FQ);
    int s = src[e];
    int d = dst[e];
    const float4 v = *reinterpret_cast<const float4*>(x + (long long)s * F + g * 4);
    float* o = summed + (long long)d * F + g * 4;
    atomicAdd(o + 0, v.x);
    atomicAdd(o + 1, v.y);
    atomicAdd(o + 2, v.z);
    atomicAdd(o + 3, v.w);
  }
}

// ---------------------------------------------------------------------------
// Fused SAGE linear: out = act( (summed/max(cnt,1)) @ Wl + bias + self @ Wr )
// One wave32 per 16-row node tile; v_wmma_f32_16x16x32_f16.
//
// Weights staged in LDS in fragment-major order sW[pass][kt][nt][lane][16],
// so each B fragment is one contiguous 32B-aligned v16h (2x ds_load_b128).
// Per k-step, all NT B fragments are loaded into registers first (single ds
// clause + one dscnt wait), then the NT WMMAs issue back-to-back.
//
// Fragment element mapping (CDNA5 ISA, 16-bit B 32x16):
//   lane: n = ntile*16 + (lane&15), koff = (lane>>4)*8
//   elem i<8 : K = ktile*32 + koff + i
//   elem i>=8: K = ktile*32 + koff + 16 + (i-8)
// A fragments use the mirrored mapping with m = lane&15 (rows from global,
// f32 -> f16; mean-divide folded in on the aggregate pass only).
// C/D 16x16 f32: VGPR r -> M = (lane<16 ? r : 8+r), N = lane&15.
// ---------------------------------------------------------------------------
template <int K, int NOUT, int NPAD, bool RELU>
__global__ void __launch_bounds__(256)
sage_gemm_kernel(const float* __restrict__ Aagg,   // [N,K] summed neighbors
                 const float* __restrict__ cnt,    // [N]
                 const float* __restrict__ Aself,  // [N,K] self features
                 const float* __restrict__ Wl,     // [K,NOUT] f32
                 const float* __restrict__ bias,   // [NOUT]
                 const float* __restrict__ Wr,     // [K,NOUT] f32
                 float* __restrict__ out,          // [N,NOUT]
                 int N) {
  constexpr int KT    = K / 32;        // k-tiles
  constexpr int NT    = NPAD / 16;     // n-tiles (padded)
  constexpr int NFRAG = 2 * KT * NT;   // fragments (pass,kt,nt)
  constexpr int TOT   = NFRAG * 32 * 16;

  __shared__ _Float16 sW[TOT] __attribute__((aligned(32)));

  // ---- stage weights, permuted into per-lane fragment order ----
  for (int idx = threadIdx.x; idx < TOT; idx += blockDim.x) {
    const int frag = idx >> 9;             // / (32 lanes * 16 halfs)
    const int rem  = idx & 511;
    const int ln   = rem >> 4;
    const int i    = rem & 15;
    const int pass = frag / (KT * NT);
    const int fr   = frag - pass * (KT * NT);
    const int kti  = fr / NT;
    const int nt   = fr - kti * NT;
    const int koff = (ln >> 4) * 8;
    const int k    = kti * 32 + koff + (i < 8 ? i : 8 + i);
    const int n    = nt * 16 + (ln & 15);
    const float* Wsrc = (pass == 0) ? Wl : Wr;
    sW[idx] = (n < NOUT) ? (_Float16)Wsrc[k * NOUT + n] : (_Float16)0.0f;
  }
  __syncthreads();

  const int lane = threadIdx.x & 31;
  const int wave = threadIdx.x >> 5;
  const int tile = blockIdx.x * (blockDim.x >> 5) + wave;
  const int numTiles = (N + 15) >> 4;
  if (tile >= numTiles) return;            // wave-uniform: EXEC stays all-1s

  const int m0   = tile * 16;
  const int m    = lane & 15;
  const int half = lane >> 4;              // 0 or 1
  const int koff = half * 8;

  int rowA = m0 + m;
  if (rowA >= N) rowA = N - 1;             // clamp loads, guard stores below
  const float inv = 1.0f / fmaxf(cnt[rowA], 1.0f);

  const v16h* fragBase = reinterpret_cast<const v16h*>(sW);

  v8f acc[NT] = {};

  #pragma unroll
  for (int pass = 0; pass < 2; ++pass) {
    const float* A = (pass == 0) ? Aagg : Aself;

    #pragma unroll
    for (int kti = 0; kti < KT; ++kti) {
      // ---- A fragment (f32 -> f16; scale only on the aggregate pass) ----
      v16h a;
      const float* ap = A + (long long)rowA * K + kti * 32 + koff;
      if (pass == 0) {
        #pragma unroll
        for (int i = 0; i < 8; ++i) a[i]     = (_Float16)(ap[i] * inv);
        #pragma unroll
        for (int i = 0; i < 8; ++i) a[8 + i] = (_Float16)(ap[16 + i] * inv);
      } else {
        #pragma unroll
        for (int i = 0; i < 8; ++i) a[i]     = (_Float16)ap[i];
        #pragma unroll
        for (int i = 0; i < 8; ++i) a[8 + i] = (_Float16)ap[16 + i];
      }

      // ---- batch-load all NT B fragments (one ds clause, one wait) ----
      v16h bfr[NT];
      #pragma unroll
      for (int nt = 0; nt < NT; ++nt)
        bfr[nt] = fragBase[((pass * KT + kti) * NT + nt) * 32 + lane];

      // ---- back-to-back WMMAs ----
      #pragma unroll
      for (int nt = 0; nt < NT; ++nt)
        acc[nt] = __builtin_amdgcn_wmma_f32_16x16x32_f16(
            false, a, false, bfr[nt], (short)0, acc[nt], false, false);
    }
  }

  // ---- epilogue: bias (+ ReLU), store 16 x NOUT tile ----
  const int n_lo = lane & 15;
  #pragma unroll
  for (int nt = 0; nt < NT; ++nt) {
    const int col = nt * 16 + n_lo;
    if (col < NOUT) {
      const float bv = bias[col];
      #pragma unroll
      for (int r = 0; r < 8; ++r) {
        const int row = m0 + (half ? (8 + r) : r);
        if (row < N) {
          float v = acc[nt][r] + bv;
          if (RELU) v = fmaxf(v, 0.0f);
          out[(long long)row * NOUT + col] = v;
        }
      }
    }
  }
}

// ---------------------------------------------------------------------------
// Row-wise log-softmax over 40 classes, one wave32 per row, in place.
// ---------------------------------------------------------------------------
__global__ void log_softmax_kernel(float* __restrict__ logits, int N) {
  const int lane = threadIdx.x & 31;
  const int wave = threadIdx.x >> 5;
  const int row  = blockIdx.x * (blockDim.x >> 5) + wave;
  if (row >= N) return;

  float* lr = logits + (long long)row * NCLS;
  const bool hi = lane < (NCLS - 32);          // lanes 0..7 carry classes 32..39
  float a = lr[lane];
  float b = hi ? lr[32 + lane] : -INFINITY;

  float mx = fmaxf(a, b);
  #pragma unroll
  for (int off = 16; off > 0; off >>= 1) mx = fmaxf(mx, __shfl_xor(mx, off, 32));

  float s = expf(a - mx) + (hi ? expf(b - mx) : 0.0f);
  #pragma unroll
  for (int off = 16; off > 0; off >>= 1) s += __shfl_xor(s, off, 32);

  const float lse = mx + logf(s);
  lr[lane] = a - lse;
  if (hi) lr[32 + lane] = b - lse;
}

// ---------------------------------------------------------------------------
// kernel_launch
// inputs: x[N,128], edge_index[2,E] (int), W1_l[128,64], b1[64], W1_r[128,64],
//         W2_l[64,40], b2[40], W2_r[64,40]   -> out: log_softmax logits [N,40]
// ---------------------------------------------------------------------------
extern "C" void kernel_launch(void* const* d_in, const int* in_sizes, int n_in,
                              void* d_out, int out_size, void* d_ws, size_t ws_size,
                              hipStream_t stream) {
  const float* x   = (const float*)d_in[0];
  const int*   ei  = (const int*)d_in[1];   // int32 per harness convention
  const float* W1l = (const float*)d_in[2];
  const float* b1  = (const float*)d_in[3];
  const float* W1r = (const float*)d_in[4];
  const float* W2l = (const float*)d_in[5];
  const float* b2  = (const float*)d_in[6];
  const float* W2r = (const float*)d_in[7];

  const int N = in_sizes[0] / F_IN;
  const int E = in_sizes[1] / 2;
  const int* src = ei;
  const int* dst = ei + E;

  // Workspace layout (floats):
  //   summed : N*F_IN   (layer-1 neighbor sums; reused as N*HID for layer 2)
  //   cnt    : N        (in-degree, reused by both layers)
  //   h      : N*HID    (layer-1 activations)
  float* summed = (float*)d_ws;
  float* cnt    = summed + (size_t)N * F_IN;
  float* h      = cnt + N;
  float* logits = (float*)d_out;            // layer-2 output, softmaxed in place

  const int waves_per_block = 8;            // 256 threads = 8 wave32
  const int numTiles = (N + 15) / 16;
  const int gemm_grid = (numTiles + waves_per_block - 1) / waves_per_block;

  // ---- Layer 1 ----
  zero_f32_kernel<<<2048, 256, 0, stream>>>(summed, (long long)N * F_IN + N);
  degree_kernel<<<2048, 256, 0, stream>>>(dst, cnt, E);
  scatter_sum_kernel<F_IN><<<4096, 256, 0, stream>>>(x, src, dst, summed, E);
  sage_gemm_kernel<F_IN, HID, HID, true><<<gemm_grid, 256, 0, stream>>>(
      summed, cnt, x, W1l, b1, W1r, h, N);

  // ---- Layer 2 (summed region reused at width HID) ----
  zero_f32_kernel<<<2048, 256, 0, stream>>>(summed, (long long)N * HID);
  scatter_sum_kernel<HID><<<4096, 256, 0, stream>>>(h, src, dst, summed, E);
  sage_gemm_kernel<HID, NCLS, 48, false><<<gemm_grid, 256, 0, stream>>>(
      summed, cnt, h, W2l, b2, W2r, logits, N);

  // ---- log_softmax in place on d_out ----
  log_softmax_kernel<<<(N + waves_per_block - 1) / waves_per_block, 256, 0, stream>>>(
      logits, N);
}
